// NearestEmbed_11218454577359
// MI455X (gfx1250) — matrix operand order repaired and checked by reading, not compiled, not run on hardware
//
#include <hip/hip_runtime.h>
#include <hip/hip_bf16.h>

// MI455X / gfx1250, wave32. WMMA f32<=f16 16x16x32 + TDM async tile staging.
typedef __attribute__((ext_vector_type(16))) _Float16 v16h;
typedef __attribute__((ext_vector_type(8)))  float    v8f;
typedef unsigned int u32x4 __attribute__((ext_vector_type(4)));
typedef int          i32x8 __attribute__((ext_vector_type(8)));
typedef int          i32x4 __attribute__((ext_vector_type(4)));

#define D_DIM 256
#define K_DIM 512
#define HW    1024
#define B_DIM 32
#define M_TOT (B_DIM * HW)          // 32768 pixel vectors
#define N_TILES (K_DIM / 16)        // 32 code tiles
#define TILE_HALFS 4096             // f16 elements per hi (or lo) tile = 256*16
#define TILE_BYTES 16384            // hi + lo tile, bytes

// ---------------------------------------------------------------------------
// Kernel 1: e2[k] = sum_d w[d,k]^2
// ---------------------------------------------------------------------------
__global__ void vq_e2_kernel(const float* __restrict__ w, float* __restrict__ e2) {
    int k = blockIdx.x * blockDim.x + threadIdx.x;
    if (k >= K_DIM) return;
    float s = 0.0f;
#pragma unroll 4
    for (int d = 0; d < D_DIM; ++d) {
        float v = w[(size_t)d * K_DIM + k];
        s += v * v;
    }
    e2[k] = s;
}

// ---------------------------------------------------------------------------
// Kernel 2: pre-convert weight f32 -> split f16 (hi, lo) and pre-swizzle into
// the exact WMMA B-fragment layout, one 16KB contiguous block per code tile:
//   element (d, col) -> halfword index (d>>4)*256 + col*16 + (d&15)
// so the hot loop's staging is a pure linear copy (TDM-friendly).
// ---------------------------------------------------------------------------
__global__ void vq_prep_kernel(const float* __restrict__ w, _Float16* __restrict__ wswz) {
    const int nt = blockIdx.x;       // tile
    const int d  = threadIdx.x;      // 0..255
    const int k0 = nt * 16;
    const float4* wrow = (const float4*)(w + (size_t)d * K_DIM + k0);
    _Float16* dh = wswz + (size_t)nt * (2 * TILE_HALFS) + ((d >> 4) * 256 + (d & 15));
    _Float16* dl = dh + TILE_HALFS;
#pragma unroll
    for (int q = 0; q < 4; ++q) {
        float4 f = wrow[q];
        float vv[4] = { f.x, f.y, f.z, f.w };
#pragma unroll
        for (int r = 0; r < 4; ++r) {
            _Float16 h = (_Float16)vv[r];
            dh[(q * 4 + r) * 16] = h;
            dl[(q * 4 + r) * 16] = (_Float16)(vv[r] - (float)h);  // residual
        }
    }
}

// ---------------------------------------------------------------------------
// TDM: async copy one 16KB pre-swizzled tile (global -> LDS).
// D# built per CDNA5 ISA ch.8: data_size=8B, 1D copy of 2048 elements.
// ---------------------------------------------------------------------------
__device__ __forceinline__ void tdm_load_tile(const _Float16* gsrc, unsigned lds_off) {
    unsigned long long ga = (unsigned long long)(uintptr_t)gsrc;
    u32x4 g0 = { 1u,                                   // count=1, user-mode D#
                 lds_off,                              // lds_addr (bytes)
                 (unsigned)ga,                         // global_addr[31:0]
                 (unsigned)((ga >> 32) & 0x1FFFFFFu) | (2u << 30) }; // [56:32] | type=2
    i32x8 g1 = { (int)(3u << 16),        // wg_mask=0, data_size=3 (8B)
                 (int)(2048u << 16),     // tensor_dim0[15:0]=2048
                 (int)(1u << 16),        // tensor_dim0 hi=0 | tensor_dim1[15:0]=1
                 (int)(2048u << 16),     // tensor_dim1 hi=0 | tile_dim0=2048
                 1,                      // tile_dim1=1 | tile_dim2=0
                 2048,                   // tensor_dim0_stride[31:0]
                 (int)(2048u << 16),     // t0s hi=0 | tensor_dim1_stride[15:0]=2048
                 0 };
    i32x4 g2 = { 0, 0, 0, 0 };
    i32x4 g3 = { 0, 0, 0, 0 };
#if __clang_major__ >= 23
    i32x8 g4 = { 0, 0, 0, 0, 0, 0, 0, 0 };
    __builtin_amdgcn_tensor_load_to_lds(g0, g1, g2, g3, g4, 0);
#else
    __builtin_amdgcn_tensor_load_to_lds(g0, g1, g2, g3, 0);
#endif
}

// ---------------------------------------------------------------------------
// Kernel 3: fused cross-GEMM (split-f16 WMMA) + argmin.
//   d2(m,k) = e2[k] - 2 * <x_m, w_k>   (x^2 term constant in k -> dropped)
//   Weight tiles double-buffered in LDS via TDM; compute overlaps next DMA.
// ---------------------------------------------------------------------------
__launch_bounds__(256, 1)
__global__ void vq_argmin_wmma(const float* __restrict__ x,
                               const _Float16* __restrict__ wtiles,
                               const float* __restrict__ e2g,
                               int* __restrict__ argmin_out) {
    __shared__ _Float16 s_wt[2][2 * TILE_HALFS];  // [buf][hi(4096) | lo(4096)]
    __shared__ float    s_e2[K_DIM];

    const int tid  = threadIdx.x;
    const int lane = tid & 31;
    const int wave = tid >> 5;
    const int col  = lane & 15;   // C/D column (N) and A row (M) within tile
    const int half = lane >> 4;

    s_e2[tid]       = e2g[tid];
    s_e2[tid + 256] = e2g[tid + 256];

    // ---- A tile (16 pixels x 256 d) in registers, split f32 -> f16 hi/lo.
    // A 16x32 layout used here (contiguous k, same as B): row M = lane%16; element j:
    //   K = 16*half + j within the 32-wide chunk.
    const int m  = blockIdx.x * 128 + wave * 16 + col;
    const int b  = m >> 10;
    const int n  = m & (HW - 1);
    const float* xb = x + (size_t)b * D_DIM * HW + n;

    v16h a_hi[8], a_lo[8];
#pragma unroll
    for (int c = 0; c < 8; ++c) {
#pragma unroll
        for (int j = 0; j < 16; ++j) {
            int d = c * 32 + 16 * half + j;
            float v = xb[(size_t)d * HW];
            _Float16 h = (_Float16)v;
            a_hi[c][j] = h;
            a_lo[c][j] = (_Float16)(v - (float)h);
        }
    }

    float best[8];
    int   bidx[8];
#pragma unroll
    for (int i = 0; i < 8; ++i) { best[i] = 3.4e38f; bidx[i] = 0; }

    // preload tile 0 (TDM ignores EXEC; wave0 issues one DMA)
    if (wave == 0) tdm_load_tile(wtiles, (unsigned)(uintptr_t)&s_wt[0][0]);

    for (int nt = 0; nt < N_TILES; ++nt) {
        __syncthreads();   // prior compute done: buf[(nt+1)&1] is free to overwrite
        if (wave == 0) {
            if (nt + 1 < N_TILES) {
                tdm_load_tile(wtiles + (size_t)(nt + 1) * (2 * TILE_HALFS),
                              (unsigned)(uintptr_t)&s_wt[(nt + 1) & 1][0]);
                __builtin_amdgcn_s_wait_tensorcnt(1);  // in-order: tile nt complete
            } else {
                __builtin_amdgcn_s_wait_tensorcnt(0);
            }
        }
        __syncthreads();   // buf[nt&1] ready for all waves

        const _Float16* bufh = &s_wt[nt & 1][0];
        v8f acc = {};
#pragma unroll
        for (int c = 0; c < 8; ++c) {
            const int off = (2 * c + half) * 256 + col * 16;   // 32B-aligned
            v16h bh = *(const v16h*)&bufh[off];
            v16h bl = *(const v16h*)&bufh[TILE_HALFS + off];
            // split product: hi*hi + hi*lo + lo*hi (lo*lo ~ 2^-24, dropped)
            acc = __builtin_amdgcn_wmma_f32_16x16x32_f16(false, a_hi[c], false, bh,
                                                         (short)0, acc, false, false);
            acc = __builtin_amdgcn_wmma_f32_16x16x32_f16(false, a_hi[c], false, bl,
                                                         (short)0, acc, false, false);
            acc = __builtin_amdgcn_wmma_f32_16x16x32_f16(false, a_lo[c], false, bh,
                                                         (short)0, acc, false, false);
        }

        // C layout: lane col = N, VGPR i = row M = i + 8*half
        const int   k0   = nt * 16;
        const float e2v  = s_e2[k0 + col];
        const int   kidx = k0 + col;
#pragma unroll
        for (int i = 0; i < 8; ++i) {
            float dist = e2v - 2.0f * acc[i];
            if (dist < best[i]) { best[i] = dist; bidx[i] = kidx; }  // strict <: first min
        }
    }

    // ---- cross-lane argmin reduction within each 16-lane column group
#pragma unroll
    for (int mask = 1; mask <= 8; mask <<= 1) {
#pragma unroll
        for (int i = 0; i < 8; ++i) {
            float ov = __shfl_xor(best[i], mask, 32);
            int   oi = __shfl_xor(bidx[i], mask, 32);
            if (ov < best[i] || (ov == best[i] && oi < bidx[i])) {
                best[i] = ov; bidx[i] = oi;   // tie -> smaller k (jnp.argmin semantics)
            }
        }
    }

    if (col == 0) {
        const int mrow = blockIdx.x * 128 + wave * 16 + half * 8;
#pragma unroll
        for (int i = 0; i < 8; ++i) argmin_out[mrow + i] = bidx[i];
    }
}

// ---------------------------------------------------------------------------
// Kernel 4: gather winning codebook columns into (B, D, H, W) output.
// ---------------------------------------------------------------------------
__global__ void vq_gather_kernel(const float* __restrict__ w,
                                 const int* __restrict__ argmin,
                                 float* __restrict__ out) {
    const int t  = threadIdx.x;
    const int m  = blockIdx.x * 256 + t;
    const int k  = argmin[m];
    const int b  = m >> 10;
    const int n  = m & (HW - 1);
    float* ob       = out + (size_t)b * D_DIM * HW + n;
    const float* wc = w + k;
#pragma unroll 4
    for (int d = 0; d < D_DIM; ++d) {
        __builtin_nontemporal_store(wc[(size_t)d * K_DIM], &ob[(size_t)d * HW]);
    }
}

// ---------------------------------------------------------------------------
extern "C" void kernel_launch(void* const* d_in, const int* in_sizes, int n_in,
                              void* d_out, int out_size, void* d_ws, size_t ws_size,
                              hipStream_t stream) {
    const float* x = (const float*)d_in[0];   // (32, 256, 32, 32) f32
    const float* w = (const float*)d_in[1];   // (256, 512) f32

    float* out_res = (float*)d_out;                                      // B*D*HW f32
    int*   out_idx = (int*)((float*)d_out + (size_t)B_DIM * D_DIM * HW); // B*HW i32

    _Float16* wswz = (_Float16*)d_ws;                                    // 512 KB tiles
    float*    e2   = (float*)((char*)d_ws + (size_t)N_TILES * TILE_BYTES); // +2 KB

    vq_e2_kernel  <<<2,           256, 0, stream>>>(w, e2);
    vq_prep_kernel<<<N_TILES,     256, 0, stream>>>(w, wswz);
    vq_argmin_wmma<<<M_TOT / 128, 256, 0, stream>>>(x, wswz, e2, out_idx);
    vq_gather_kernel<<<M_TOT / 256, 256, 0, stream>>>(w, out_idx, out_res);
}
